// GroupedQueryAttention_48120813584715
// MI455X (gfx1250) — matrix operand order repaired and checked by reading, not compiled, not run on hardware
//
#include <hip/hip_runtime.h>
#include <hip/hip_bf16.h>

typedef __bf16 bf16;
typedef __attribute__((ext_vector_type(16))) __bf16 bf16x16;
typedef __attribute__((ext_vector_type(8)))  __bf16 bf16x8;
typedef __attribute__((ext_vector_type(4)))  __bf16 bf16x4;
typedef __attribute__((ext_vector_type(8)))  float   f32x8;
typedef __attribute__((ext_vector_type(4)))  unsigned int u32x4;
typedef __attribute__((ext_vector_type(8)))  unsigned int u32x8;

#define NHEADS 16
#define NKV    4
#define HD     128
#define TSEQ   2048
#define CDIM   2048
#define KVDIM  512
#define BATCH  4

static __device__ inline f32x8 wmma_bf16(bf16x16 a, bf16x16 b, f32x8 c) {
  // v_wmma_f32_16x16x32_bf16: D = A(16x32) * B(32x16) + C(16x16 f32)
  return __builtin_amdgcn_wmma_f32_16x16x32_bf16(false, a, false, b, (short)0, c,
                                                 false, false);
}

static __device__ inline f32x8 zero8() {
  f32x8 z;
#pragma unroll
  for (int i = 0; i < 8; i++) z[i] = 0.0f;
  return z;
}

// A-fragment (16x32, 16-bit): lane<16 -> row M=lane, elems K0..7 | K16..23
//                             lane>=16 -> row M=lane-16, elems K8..15 | K24..31
static __device__ inline bf16x16 load_fragA(const bf16* __restrict__ p,
                                            int row0, int k0, int ld) {
  const int lane = threadIdx.x & 31;
  const bf16* base =
      p + (size_t)(row0 + (lane & 15)) * ld + k0 + ((lane >> 4) << 3);
  bf16x8 lo = *(const bf16x8*)(base);
  bf16x8 hi = *(const bf16x8*)(base + 16);
  return __builtin_shufflevector(lo, hi, 0, 1, 2, 3, 4, 5, 6, 7,
                                 8, 9, 10, 11, 12, 13, 14, 15);
}

// B-fragment (32x16, 16-bit), source stored as rows of N with contiguous K:
// lane<16 -> col N=lane, elems K0..7 | K16..23 ; lane>=16 -> col N=lane-16, K8..15 | K24..31
static __device__ inline bf16x16 load_fragB(const bf16* __restrict__ p,
                                            int row0, int k0, int ld) {
  const int lane = threadIdx.x & 31;
  const bf16* base =
      p + (size_t)(row0 + (lane & 15)) * ld + k0 + ((lane >> 4) << 3);
  bf16x8 lo = *(const bf16x8*)(base);
  bf16x8 hi = *(const bf16x8*)(base + 16);
  return __builtin_shufflevector(lo, hi, 0, 1, 2, 3, 4, 5, 6, 7,
                                 8, 9, 10, 11, 12, 13, 14, 15);
}

// Same fragment shapes but sourced from an LDS tile with row stride 64 elems.
static __device__ inline bf16x16 fragA_lds(const bf16* base, int row0, int ks) {
  const int lane = threadIdx.x & 31;
  const bf16* p = base + (row0 + (lane & 15)) * 64 + ks + ((lane >> 4) << 3);
  bf16x8 lo = *(const bf16x8*)(p);
  bf16x8 hi = *(const bf16x8*)(p + 16);
  return __builtin_shufflevector(lo, hi, 0, 1, 2, 3, 4, 5, 6, 7,
                                 8, 9, 10, 11, 12, 13, 14, 15);
}
static __device__ inline bf16x16 fragB_lds(const bf16* base, int row0, int ks) {
  const int lane = threadIdx.x & 31;
  const bf16* p = base + (row0 + (lane & 15)) * 64 + ks + ((lane >> 4) << 3);
  bf16x8 lo = *(const bf16x8*)(p);
  bf16x8 hi = *(const bf16x8*)(p + 16);
  return __builtin_shufflevector(lo, hi, 0, 1, 2, 3, 4, 5, 6, 7,
                                 8, 9, 10, 11, 12, 13, 14, 15);
}

// Async copy of one 16B chunk global -> LDS (tracked by ASYNCcnt).
static __device__ inline void async_b128(const void* gaddr, unsigned lds_off) {
  asm volatile("global_load_async_to_lds_b128 %0, %1, off"
               :: "v"(lds_off), "v"(gaddr)
               : "memory");
}

// TDM: DMA a 2D tile (rows x cols bf16, row stride ld elements) into LDS.
// Builds the D# descriptor per CDNA5 ISA 8.3/8.4 and issues TENSOR_LOAD_TO_LDS
// (tracked by TENSORcnt). Issue from one wave; EXEC is ignored by TDM.
static __device__ inline void tdm_load_tile(const bf16* gsrc, const bf16* ldst,
                                            unsigned rows, unsigned cols,
                                            unsigned ld) {
  const unsigned long long ga = (unsigned long long)(size_t)gsrc;
  u32x4 g0;
  g0[0] = 1u;                                        // count=1, user D#
  g0[1] = (unsigned)(size_t)ldst;                    // lds_addr (bytes)
  g0[2] = (unsigned)ga;                              // global_addr[31:0]
  g0[3] = (unsigned)((ga >> 32) & 0x01FFFFFFu) | (2u << 30);  // addr[56:32]|type=2
  u32x8 g1;
  g1[0] = 1u << 16;                // wg_mask=0, data_size=1 (2 bytes)
  g1[1] = (cols & 0xFFFFu) << 16;  // tensor_dim0[15:0]
  g1[2] = (cols >> 16) | ((rows & 0xFFFFu) << 16);   // dim0[31:16] | dim1[15:0]
  g1[3] = (rows >> 16) | ((cols & 0xFFFFu) << 16);   // dim1[31:16] | tile_dim0
  g1[4] = rows & 0xFFFFu;          // tile_dim1 | tile_dim2=0
  g1[5] = ld;                      // tensor_dim0_stride[31:0]
  g1[6] = 0u;                      // stride0[47:32] | stride1[15:0]
  g1[7] = 0u;                      // stride1[47:16]
  asm volatile("tensor_load_to_lds %0, %1" :: "s"(g0), "s"(g1) : "memory");
}

// ---------------- fp32 -> bf16 conversion ----------------
__global__ void __launch_bounds__(256)
cvt_f32_bf16(const float* __restrict__ in, bf16* __restrict__ out, size_t n) {
  size_t i = ((size_t)blockIdx.x * blockDim.x + threadIdx.x) * 4;
  if (i + 3 < n) {
    float4 v = *(const float4*)(in + i);
    bf16x4 o;
    o[0] = (bf16)v.x; o[1] = (bf16)v.y; o[2] = (bf16)v.z; o[3] = (bf16)v.w;
    *(bf16x4*)(out + i) = o;
  }
}

// ---------------- WMMA GEMM:  Y = A(MxK) @ W(NxK)^T ----------------
// MODE 0: bf16 out, row-major [M x N]
// MODE 1: bf16 out, per-batch transposed:  out[(b*KVDIM + n)*TSEQ + t]
// MODE 2: f32  out, row-major [M x N]
// Block: 256 threads = 8 waves; block tile 128(M) x 128(N); wave tile 32x64.
// K staged 64-deep, double buffered: A tile via per-lane async-to-LDS copies,
// B tile via one TDM descriptor (wave 0), both overlapped with WMMA compute.
template <int MODE>
__global__ void __launch_bounds__(256)
gemm_bf16_wmma(const bf16* __restrict__ A, const bf16* __restrict__ W,
               void* __restrict__ out, int N, int K) {
  __shared__ alignas(16) bf16 sA[2][128 * 64];  // 16KB per buffer
  __shared__ alignas(16) bf16 sB[2][128 * 64];

  const int tid = threadIdx.x;
  const int lane = tid & 31;
  const int wave = tid >> 5;
  const int mBlk = blockIdx.y * 128;
  const int nBlk = blockIdx.x * 128;
  const int wm = (wave & 3) * 32;   // wave M offset within block
  const int wn = (wave >> 2) * 64;  // wave N offset within block
  const bool tdmWave = (wave == 0);

  f32x8 acc[2][4];
#pragma unroll
  for (int i = 0; i < 2; i++)
#pragma unroll
    for (int j = 0; j < 4; j++) acc[i][j] = zero8();

  // A tile (128x64) via async-VMEM: 4 x 16B chunks per thread.
  auto stageA = [&](int buf, int k0) {
#pragma unroll
    for (int i = 0; i < 4; i++) {
      const int c = tid + 256 * i;       // 1024 chunks of 16B per tile
      const int row = c >> 3;            // 8 chunks per 64-elem row
      const int col = (c & 7) * 8;       // element offset within row
      async_b128(A + (size_t)(mBlk + row) * K + k0 + col,
                 (unsigned)(size_t)(&sA[buf][c * 8]));
    }
  };
  // B tile (128x64) via one TDM 2D descriptor.
  auto stageB = [&](int buf, int k0) {
    tdm_load_tile(W + (size_t)nBlk * K + k0, &sB[buf][0], 128u, 64u,
                  (unsigned)K);
  };

  stageA(0, 0);
  if (tdmWave) stageB(0, 0);
  const int nstage = K / 64;
  for (int s = 0; s < nstage; s++) {
    // Drain my async loads (A) and, on wave 0, the TDM transfer (B).
    asm volatile("s_wait_asynccnt 0x0" ::: "memory");
    if (tdmWave) asm volatile("s_wait_tensorcnt 0x0" ::: "memory");
    // Publish LDS to all waves; also: everyone done reading the refill buffer.
    __syncthreads();
    if (s + 1 < nstage) {
      stageA((s + 1) & 1, (s + 1) * 64);            // overlaps compute
      if (tdmWave) stageB((s + 1) & 1, (s + 1) * 64);
    }

    const bf16* aBuf = sA[s & 1];
    const bf16* bBuf = sB[s & 1];
#pragma unroll
    for (int ks = 0; ks < 64; ks += 32) {
      bf16x16 a0 = fragA_lds(aBuf, wm, ks);
      bf16x16 a1 = fragA_lds(aBuf, wm + 16, ks);
#pragma unroll
      for (int ni = 0; ni < 4; ni++) {
        bf16x16 b = fragB_lds(bBuf, wn + ni * 16, ks);
        acc[0][ni] = wmma_bf16(a0, b, acc[0][ni]);
        acc[1][ni] = wmma_bf16(a1, b, acc[1][ni]);
      }
    }
  }

  const int l15 = lane & 15;
  const int hl8 = (lane >> 4) << 3;  // 0 or 8 (M offset in C/D layout)
#pragma unroll
  for (int mi = 0; mi < 2; mi++)
#pragma unroll
    for (int ni = 0; ni < 4; ni++) {
      const int mt = mBlk + wm + mi * 16;
      const int nc = nBlk + wn + ni * 16 + l15;
      if (MODE == 0) {
        bf16* o = (bf16*)out;
#pragma unroll
        for (int r = 0; r < 8; r++)
          o[(size_t)(mt + r + hl8) * N + nc] = (bf16)acc[mi][ni][r];
      } else if (MODE == 2) {
        float* o = (float*)out;
#pragma unroll
        for (int r = 0; r < 8; r++)
          o[(size_t)(mt + r + hl8) * N + nc] = acc[mi][ni][r];
      } else {  // MODE 1: transposed V store, contiguous in t -> 16B store
        const int bidx = mt / TSEQ;
        const int t0 = (mt % TSEQ) + hl8;
        bf16x8 v;
#pragma unroll
        for (int r = 0; r < 8; r++) v[r] = (bf16)acc[mi][ni][r];
        *(bf16x8*)((bf16*)out + ((size_t)bidx * KVDIM + nc) * TSEQ + t0) = v;
      }
    }
}

// ---------------- Flash attention (causal, GQA) ----------------
// One wave per (batch, head, 16-query tile). Q:[B*T,C]  K:[B*T,KV]  VT:[B,KV,T]
__global__ void __launch_bounds__(32)
gqa_flash(const bf16* __restrict__ Q, const bf16* __restrict__ Kb,
          const bf16* __restrict__ VT, bf16* __restrict__ O) {
  __shared__ bf16 Pl[16][40];  // 16 rows x 32 keys, padded

  const int lane = threadIdx.x & 31;
  const int l15 = lane & 15;
  const int hl8 = (lane >> 4) << 3;
  const int qt = blockIdx.x, h = blockIdx.y, b = blockIdx.z;
  const int kvh = h >> 2;  // n_rep = 4
  const int qrow0 = b * TSEQ + qt * 16;
  const int qcol = h * HD;
  const int kcol = kvh * HD;
  const int vrow0 = b * KVDIM + kvh * HD;
  const float scale = 0.08838834764831845f;  // 1/sqrt(128)

  bf16x16 qf[4];
#pragma unroll
  for (int i = 0; i < 4; i++) qf[i] = load_fragA(Q, qrow0, qcol + i * 32, CDIM);

  f32x8 oacc[8];
#pragma unroll
  for (int t = 0; t < 8; t++) oacc[t] = zero8();
  float mi[8], li[8];
#pragma unroll
  for (int r = 0; r < 8; r++) { mi[r] = -1e30f; li[r] = 0.0f; }

  const int kmax = (qt * 16 + 15) >> 5;
  for (int kt = 0; kt <= kmax; kt++) {
    const int key0 = kt * 32;
    // ---- S = Q @ K^T over head dim (4 k-steps), two 16-key tiles ----
    f32x8 s0 = zero8(), s1 = zero8();
#pragma unroll
    for (int kk = 0; kk < 4; kk++) {
      bf16x16 k0f = load_fragB(Kb, b * TSEQ + key0, kcol + kk * 32, KVDIM);
      bf16x16 k1f = load_fragB(Kb, b * TSEQ + key0 + 16, kcol + kk * 32, KVDIM);
      s0 = wmma_bf16(qf[kk], k0f, s0);
      s1 = wmma_bf16(qf[kk], k1f, s1);
    }
    // ---- online softmax (row stats across 16-lane groups) ----
    float alpha[8];
#pragma unroll
    for (int r = 0; r < 8; r++) {
      const int qv = qt * 16 + r + hl8;
      float a0 = (key0 + l15 <= qv) ? s0[r] * scale : -1e30f;
      float a1 = (key0 + 16 + l15 <= qv) ? s1[r] * scale : -1e30f;
      float rm = fmaxf(a0, a1);
#pragma unroll
      for (int msk = 1; msk < 16; msk <<= 1)
        rm = fmaxf(rm, __shfl_xor(rm, msk, 32));
      const float mnew = fmaxf(mi[r], rm);
      const float p0 = __expf(a0 - mnew);
      const float p1 = __expf(a1 - mnew);
      float rs = p0 + p1;
#pragma unroll
      for (int msk = 1; msk < 16; msk <<= 1) rs += __shfl_xor(rs, msk, 32);
      const float al = __expf(mi[r] - mnew);
      li[r] = li[r] * al + rs;
      mi[r] = mnew;
      alpha[r] = al;
      Pl[r + hl8][l15] = (bf16)p0;       // C-layout -> LDS row-major
      Pl[r + hl8][16 + l15] = (bf16)p1;
    }
#pragma unroll
    for (int t = 0; t < 8; t++)
#pragma unroll
      for (int r = 0; r < 8; r++) oacc[t][r] *= alpha[r];

    // ---- P as A-fragment (16x32) from LDS ----
    bf16x16 pf;
    {
      const int kb = hl8;  // 0 or 8, matches A layout K split
#pragma unroll
      for (int i = 0; i < 8; i++) {
        pf[i] = Pl[l15][kb + i];
        pf[8 + i] = Pl[l15][kb + 16 + i];
      }
    }
    // ---- O += P @ V  (8 dim tiles), V^T gives contiguous B-fragments ----
#pragma unroll
    for (int t = 0; t < 8; t++) {
      bf16x16 vf = load_fragB(VT, vrow0 + t * 16, key0, TSEQ);
      oacc[t] = wmma_bf16(pf, vf, oacc[t]);
    }
  }

  float invl[8];
#pragma unroll
  for (int r = 0; r < 8; r++) invl[r] = 1.0f / li[r];
#pragma unroll
  for (int t = 0; t < 8; t++)
#pragma unroll
    for (int r = 0; r < 8; r++)
      O[(size_t)(qrow0 + r + hl8) * CDIM + qcol + t * 16 + l15] =
          (bf16)(oacc[t][r] * invl[r]);
}

extern "C" void kernel_launch(void* const* d_in, const int* in_sizes, int n_in,
                              void* d_out, int out_size, void* d_ws,
                              size_t ws_size, hipStream_t stream) {
  const float* x = (const float*)d_in[0];
  const float* Wq = (const float*)d_in[1];
  const float* Wk = (const float*)d_in[2];
  const float* Wv = (const float*)d_in[3];
  const float* Wo = (const float*)d_in[4];

  char* ws = (char*)d_ws;
  size_t off = 0;
  auto alloc = [&](size_t bytes) -> void* {
    void* p = ws + off;
    off += (bytes + 255) & ~(size_t)255;
    return p;
  };
  bf16* Xb  = (bf16*)alloc((size_t)BATCH * TSEQ * CDIM * 2);   // 32 MB
  bf16* Wqb = (bf16*)alloc((size_t)CDIM * CDIM * 2);           // 8 MB
  bf16* Wkb = (bf16*)alloc((size_t)KVDIM * CDIM * 2);          // 2 MB
  bf16* Wvb = (bf16*)alloc((size_t)KVDIM * CDIM * 2);          // 2 MB
  bf16* Wob = (bf16*)alloc((size_t)CDIM * CDIM * 2);           // 8 MB
  bf16* Qb  = (bf16*)alloc((size_t)BATCH * TSEQ * CDIM * 2);   // 32 MB
  bf16* Kbf = (bf16*)alloc((size_t)BATCH * TSEQ * KVDIM * 2);  // 8 MB
  bf16* VTb = (bf16*)alloc((size_t)BATCH * KVDIM * TSEQ * 2);  // 8 MB
  bf16* Ab  = (bf16*)alloc((size_t)BATCH * TSEQ * CDIM * 2);   // 32 MB

  auto cvt = [&](const float* in, bf16* out, size_t n) {
    size_t th = n / 4;
    cvt_f32_bf16<<<(unsigned)((th + 255) / 256), 256, 0, stream>>>(in, out, n);
  };
  cvt(x, Xb, (size_t)BATCH * TSEQ * CDIM);
  cvt(Wq, Wqb, (size_t)CDIM * CDIM);
  cvt(Wk, Wkb, (size_t)KVDIM * CDIM);
  cvt(Wv, Wvb, (size_t)KVDIM * CDIM);
  cvt(Wo, Wob, (size_t)CDIM * CDIM);

  const int M = BATCH * TSEQ;  // 8192
  dim3 blk(256);
  // Q = X @ Wq^T  -> [M, 2048] bf16
  gemm_bf16_wmma<0><<<dim3(CDIM / 128, M / 128), blk, 0, stream>>>(Xb, Wqb, Qb, CDIM, CDIM);
  // K = X @ Wk^T  -> [M, 512] bf16
  gemm_bf16_wmma<0><<<dim3(KVDIM / 128, M / 128), blk, 0, stream>>>(Xb, Wkb, Kbf, KVDIM, CDIM);
  // V = X @ Wv^T  -> stored transposed [B, 512, T] bf16
  gemm_bf16_wmma<1><<<dim3(KVDIM / 128, M / 128), blk, 0, stream>>>(Xb, Wvb, VTb, KVDIM, CDIM);
  // causal GQA flash attention -> Ab [M, 2048] bf16
  gqa_flash<<<dim3(TSEQ / 16, NHEADS, BATCH), dim3(32), 0, stream>>>(Qb, Kbf, VTb, Ab);
  // out = Ab @ Wo^T -> fp32
  gemm_bf16_wmma<2><<<dim3(CDIM / 128, M / 128), blk, 0, stream>>>(Ab, Wob, d_out, CDIM, CDIM);
}
